// RnnDecoder_14809047237085
// MI455X (gfx1250) — matrix operand order repaired and checked by reading, not compiled, not run on hardware
//
#include <hip/hip_runtime.h>
#include <hip/hip_bf16.h>

typedef __attribute__((ext_vector_type(16))) __bf16 v16bf;
typedef __attribute__((ext_vector_type(8)))  float  v8f;

#define HH   1024
#define BB   64
#define TT   512
#define G3   3072

// ---------------------------------------------------------------------------
// Fragment loaders per CDNA5 WMMA VGPR layouts (CDNA5 ISA §7.12.2)
// ---------------------------------------------------------------------------

// A-matrix 16x32 bf16 (MxK), row-major source, leading dim ld (elements).
// lane L: M = m0 + (L&15); K-base = k0 + (L>=16 ? 8 : 0);
// two contiguous 16-byte runs per lane -> two global_load_b128.
__device__ __forceinline__ v16bf load_a_bf16(const __bf16* __restrict__ A, int ld,
                                             int m0, int k0) {
    int lane = threadIdx.x & 31;
    const __bf16* p = A + (m0 + (lane & 15)) * ld + k0 + ((lane >> 4) << 3);
    union { v16bf v; uint4 q[2]; } u;
    const uint4* p4 = (const uint4*)p;
    u.q[0] = p4[0];   // local K 0..7   (16 bytes)
    u.q[1] = p4[2];   // local K 16..23 (byte offset 32)
    return u.v;
}

// Same A fragment but sourced from a 16x32 bf16 tile staged in LDS
// (row-major, 64 bytes per row) -> two ds_load_b128 per lane.
__device__ __forceinline__ v16bf frag_a_from_lds(const __bf16* sh) {
    int lane = threadIdx.x & 31;
    const __bf16* p = sh + (lane & 15) * 32 + ((lane >> 4) << 3);
    union { v16bf v; uint4 q[2]; } u;
    const uint4* p4 = (const uint4*)p;
    u.q[0] = p4[0];
    u.q[1] = p4[2];
    return u.v;
}

// B-matrix 32x16 bf16 (KxN) = transpose of row-major W[N_total x K_total]:
// B(k,n) = W[n*ldw + k]; two 8-bf16 runs per lane (K-base 0/8, +16) = two b128 loads.
__device__ __forceinline__ v16bf load_bT_bf16(const __bf16* __restrict__ W, int ldw,
                                              int k0, int n0) {
    int lane = threadIdx.x & 31;
    const __bf16* p = W + (n0 + (lane & 15)) * ldw + k0 + ((lane >> 4) << 3);
    union { v16bf v; uint4 q[2]; } u;
    const uint4* p4 = (const uint4*)p;
    u.q[0] = p4[0];
    u.q[1] = p4[2];
    return u.v;
}

// Stage one 16x32 bf16 A tile (1 KB) into LDS with the gfx1250 async-DMA
// path: 64 lanes each move one 16-byte chunk (tracked on ASYNCcnt).
__device__ __forceinline__ void stage_a_tile_async(const __bf16* __restrict__ src,
                                                   int m0, int kk, __bf16* dst) {
    if (threadIdx.x < 64) {
        int r = threadIdx.x >> 2;          // tile row 0..15
        int c = threadIdx.x & 3;           // 16B chunk 0..3
        unsigned voff = (unsigned)(((m0 + r) * HH + kk) * 2 + c * 16);
        unsigned lds  = (unsigned)(uintptr_t)dst + (unsigned)threadIdx.x * 16u;
        asm volatile("global_load_async_to_lds_b128 %0, %1, %2"
                     :: "v"(lds), "v"(voff), "s"(src) : "memory");
    }
}

__device__ __forceinline__ float sigmoid_f(float x) {
    return 1.0f / (1.0f + __expf(-x));
}

// ---------------------------------------------------------------------------
// Utility kernels
// ---------------------------------------------------------------------------
__global__ void f32_to_bf16_kernel(const float* __restrict__ src,
                                   __bf16* __restrict__ dst, int n) {
    int i = blockIdx.x * blockDim.x + threadIdx.x;
    if (i < n) dst[i] = (__bf16)src[i];
}

__global__ void zero_f32_kernel(float* __restrict__ p, int n) {
    int i = blockIdx.x * blockDim.x + threadIdx.x;
    if (i < n) p[i] = 0.0f;
}

__global__ void zero_bf16_kernel(__bf16* __restrict__ p, int n) {
    int i = blockIdx.x * blockDim.x + threadIdx.x;
    if (i < n) p[i] = (__bf16)0.0f;
}

// ---------------------------------------------------------------------------
// Kernel 1: x_proj[m, n] = sum_k enc[m,k] * W_ih[n,k] + b_ih[n]
// M=32768, K=1024, N=3072.  Block = 256 thr = 8 waves, all sharing one M-tile.
// A tile DMA'd into LDS (async, double-buffered); B fragments register
// ping-ponged; K-loop unrolled x2 so the two register sets alternate with
// NO cross-copy movs.
// ---------------------------------------------------------------------------
__global__ __launch_bounds__(256) void xproj_gemm_kernel(
        const __bf16* __restrict__ enc, const __bf16* __restrict__ wih,
        const float* __restrict__ bih, float* __restrict__ xp) {
    __shared__ __bf16 shA[2][16 * 32];

    int wave = threadIdx.x >> 5;
    int lane = threadIdx.x & 31;
    int m0 = blockIdx.x * 16;
    int n0 = blockIdx.y * 512 + wave * 64;

    // Prologue: stage tile kk=0, preload B(kk=0)
    stage_a_tile_async(enc, m0, 0, &shA[0][0]);
    v16bf b0[4], b1[4];
#pragma unroll
    for (int j = 0; j < 4; ++j) b0[j] = load_bT_bf16(wih, HH, 0, n0 + 16 * j);
    asm volatile("s_wait_asynccnt 0x0" ::: "memory");
    __syncthreads();
    v16bf a0 = frag_a_from_lds(&shA[0][0]);
    v16bf a1;

    v8f acc[4] = {};
    for (int kk = 0; kk < HH; kk += 64) {
        // ---- half 1: consume set0 (tile kk); prefetch tile kk+32 -> set1 ----
        stage_a_tile_async(enc, m0, kk + 32, &shA[1][0]);
#pragma unroll
        for (int j = 0; j < 4; ++j)
            b1[j] = load_bT_bf16(wih, HH, kk + 32, n0 + 16 * j);
#pragma unroll
        for (int j = 0; j < 4; ++j)
            acc[j] = __builtin_amdgcn_wmma_f32_16x16x32_bf16(
                false, a0, false, b0[j], (short)0, acc[j], false, false);
        asm volatile("s_wait_asynccnt 0x0" ::: "memory");
        __syncthreads();
        a1 = frag_a_from_lds(&shA[1][0]);

        // ---- half 2: consume set1 (tile kk+32); prefetch tile kk+64 -> set0 ----
        bool more = (kk + 64) < HH;
        if (more) {
            stage_a_tile_async(enc, m0, kk + 64, &shA[0][0]);
#pragma unroll
            for (int j = 0; j < 4; ++j)
                b0[j] = load_bT_bf16(wih, HH, kk + 64, n0 + 16 * j);
        }
#pragma unroll
        for (int j = 0; j < 4; ++j)
            acc[j] = __builtin_amdgcn_wmma_f32_16x16x32_bf16(
                false, a1, false, b1[j], (short)0, acc[j], false, false);
        if (more) {
            asm volatile("s_wait_asynccnt 0x0" ::: "memory");
            __syncthreads();
            a0 = frag_a_from_lds(&shA[0][0]);
        }
    }

    int nc  = lane & 15;
    int mro = (lane >> 4) << 3;
#pragma unroll
    for (int j = 0; j < 4; ++j) {
        int n = n0 + 16 * j + nc;
        float bias = bih[n];
#pragma unroll
        for (int v = 0; v < 8; ++v) {
            int m = m0 + v + mro;
            xp[m * G3 + n] = acc[j][v] + bias;
        }
    }
}

// ---------------------------------------------------------------------------
// Kernel 2: one GRU step.  h state lives in BOTH f32 (gate update) and bf16
// (WMMA A fragments) ping/pong buffers; K-loop unrolled x2 with two register
// sets -> pure b128-load + wmma, no copies, no conversions on the critical
// path.
// ---------------------------------------------------------------------------
__global__ __launch_bounds__(128) void gru_step_kernel(
        int t, const float* __restrict__ hprev, float* __restrict__ hnext,
        const __bf16* __restrict__ hprev_bf, __bf16* __restrict__ hnext_bf,
        const float* __restrict__ xp, const __bf16* __restrict__ whh,
        const float* __restrict__ bhh, float* __restrict__ out) {
    int wave = threadIdx.x >> 5;
    int lane = threadIdx.x & 31;
    int j0 = blockIdx.x * 64 + wave * 16;

    v16bf a0[4], b0[3], a1[4], b1[3];
#pragma unroll
    for (int mt = 0; mt < 4; ++mt)
        a0[mt] = load_a_bf16(hprev_bf, HH, mt * 16, 0);
#pragma unroll
    for (int g = 0; g < 3; ++g)
        b0[g] = load_bT_bf16(whh, HH, 0, g * HH + j0);

    v8f acc[3][4] = {};
    for (int kk = 0; kk < HH; kk += 64) {
        // ---- half 1: prefetch kk+32 into set1, compute on set0 ----
#pragma unroll
        for (int mt = 0; mt < 4; ++mt)
            a1[mt] = load_a_bf16(hprev_bf, HH, mt * 16, kk + 32);
#pragma unroll
        for (int g = 0; g < 3; ++g)
            b1[g] = load_bT_bf16(whh, HH, kk + 32, g * HH + j0);
#pragma unroll
        for (int g = 0; g < 3; ++g)
#pragma unroll
            for (int mt = 0; mt < 4; ++mt)
                acc[g][mt] = __builtin_amdgcn_wmma_f32_16x16x32_bf16(
                    false, a0[mt], false, b0[g], (short)0, acc[g][mt],
                    false, false);

        // ---- half 2: prefetch kk+64 into set0, compute on set1 ----
        bool more = (kk + 64) < HH;
        if (more) {
#pragma unroll
            for (int mt = 0; mt < 4; ++mt)
                a0[mt] = load_a_bf16(hprev_bf, HH, mt * 16, kk + 64);
#pragma unroll
            for (int g = 0; g < 3; ++g)
                b0[g] = load_bT_bf16(whh, HH, kk + 64, g * HH + j0);
        }
#pragma unroll
        for (int g = 0; g < 3; ++g)
#pragma unroll
            for (int mt = 0; mt < 4; ++mt)
                acc[g][mt] = __builtin_amdgcn_wmma_f32_16x16x32_bf16(
                    false, a1[mt], false, b1[g], (short)0, acc[g][mt],
                    false, false);
    }

    int j   = j0 + (lane & 15);
    int mro = (lane >> 4) << 3;
    float b_r = bhh[j];
    float b_z = bhh[HH + j];
    float b_n = bhh[2 * HH + j];

#pragma unroll
    for (int mt = 0; mt < 4; ++mt) {
#pragma unroll
        for (int v = 0; v < 8; ++v) {
            int brow = mt * 16 + v + mro;                 // batch index
            const float* xrow = xp + (brow * TT + t) * G3;
            float r = sigmoid_f(acc[0][mt][v] + b_r + xrow[j]);
            float z = sigmoid_f(acc[1][mt][v] + b_z + xrow[HH + j]);
            float nval = tanhf(xrow[2 * HH + j] + r * (acc[2][mt][v] + b_n));
            float hp = hprev[brow * HH + j];
            float h  = (1.0f - z) * nval + z * hp;
            hnext[brow * HH + j]    = h;
            hnext_bf[brow * HH + j] = (__bf16)h;
            out[(brow * TT + t) * HH + j] = h;
        }
    }
}

// ---------------------------------------------------------------------------
// Launch
// ---------------------------------------------------------------------------
extern "C" void kernel_launch(void* const* d_in, const int* in_sizes, int n_in,
                              void* d_out, int out_size, void* d_ws, size_t ws_size,
                              hipStream_t stream) {
    const float* enc = (const float*)d_in[0];  // [B,T,H]
    const float* Wih = (const float*)d_in[1];  // [3H,H]
    const float* Whh = (const float*)d_in[2];  // [3H,H]
    const float* bih = (const float*)d_in[3];  // [3H]
    const float* bhh = (const float*)d_in[4];  // [3H]
    float* out = (float*)d_out;                // [B,T,H]

    char* ws = (char*)d_ws;
    size_t o = 0;
    __bf16* enc_bf = (__bf16*)(ws + o); o += (size_t)BB * TT * HH * 2;   // 64 MB
    __bf16* wih_bf = (__bf16*)(ws + o); o += (size_t)G3 * HH * 2;       //  6 MB
    __bf16* whh_bf = (__bf16*)(ws + o); o += (size_t)G3 * HH * 2;       //  6 MB
    float*  xproj  = (float*)(ws + o);  o += (size_t)BB * TT * G3 * 4;  // 384 MB
    float*  hping  = (float*)(ws + o);  o += (size_t)BB * HH * 4;
    float*  hpong  = (float*)(ws + o);  o += (size_t)BB * HH * 4;
    __bf16* hbfping = (__bf16*)(ws + o); o += (size_t)BB * HH * 2;
    __bf16* hbfpong = (__bf16*)(ws + o); o += (size_t)BB * HH * 2;

    // 1) bf16 conversions (weights become L2-resident bf16; enc halves A-bytes)
    int nEnc = BB * TT * HH;
    f32_to_bf16_kernel<<<nEnc / 256, 256, 0, stream>>>(enc, enc_bf, nEnc);
    int nW = G3 * HH;
    f32_to_bf16_kernel<<<nW / 256, 256, 0, stream>>>(Wih, wih_bf, nW);
    f32_to_bf16_kernel<<<nW / 256, 256, 0, stream>>>(Whh, whh_bf, nW);

    // 2) h0 = 0 (both precisions)
    zero_f32_kernel<<<(BB * HH) / 256, 256, 0, stream>>>(hping, BB * HH);
    zero_bf16_kernel<<<(BB * HH) / 256, 256, 0, stream>>>(hbfping, BB * HH);

    // 3) batched input projection (big WMMA GEMM, async-LDS A staging)
    xproj_gemm_kernel<<<dim3((BB * TT) / 16, G3 / 512), 256, 0, stream>>>(
        enc_bf, wih_bf, bih, xproj);

    // 4) sequential scan: 512 small WMMA steps, ping/pong on h
    for (int t = 0; t < TT; ++t) {
        const float*  hp  = (t & 1) ? hpong : hping;
        float*        hn  = (t & 1) ? hping : hpong;
        const __bf16* hbp = (t & 1) ? hbfpong : hbfping;
        __bf16*       hbn = (t & 1) ? hbfping : hbfpong;
        gru_step_kernel<<<HH / 64, 128, 0, stream>>>(
            t, hp, hn, hbp, hbn, xproj, whh_bf, bhh, out);
    }
}
